// Memory_Network_5463198401218
// MI455X (gfx1250) — matrix operand order repaired and checked by reading, not compiled, not run on hardware
//
#include <hip/hip_runtime.h>
#include <math.h>

#define MEMSZ   262144
#define BATCH   256
#define DIM     512
#define CDIM    313
#define TOPK    256
#define ALPHA   0.1f
#define THRES   0.7f
#define EPSV    1e-8f
#define CAP_TIE 4096

// padded LDS row stride for the staged A matrix: 528 B = 132 DWORDs == 4 (mod 64)
// -> each lane's ds_load_b128 hits a distinct 4-bank group (conflict-optimal)
#define AROW_U4 33   // uint4s per row (528 B)

typedef __bf16 bf16x16 __attribute__((ext_vector_type(16)));
typedef float  f32x8   __attribute__((ext_vector_type(8)));
typedef float  f32x16  __attribute__((ext_vector_type(16)));

union BF16Vec { bf16x16 v; unsigned short u[16]; uint4 q[2]; };

// order-preserving float -> uint key (ascending)
static __device__ __forceinline__ unsigned int fkey(float f) {
    unsigned int u = __float_as_uint(f);
    return (u & 0x80000000u) ? ~u : (u | 0x80000000u);
}

// 64-byte async global->LDS copy (CDNA5 ASYNC path, ASYNCcnt-tracked).
// INST_OFFSET applies to both LDS and global addresses.
static __device__ __forceinline__ void async_cp64(unsigned lds, const void* g) {
    unsigned long long ga = (unsigned long long)g;
    asm volatile(
        "global_load_async_to_lds_b128 %0, %1, off\n\t"
        "global_load_async_to_lds_b128 %0, %1, off offset:16\n\t"
        "global_load_async_to_lds_b128 %0, %1, off offset:32\n\t"
        "global_load_async_to_lds_b128 %0, %1, off offset:48"
        :: "v"(lds), "v"(ga) : "memory");
}
static __device__ __forceinline__ void async_wait0() {
    asm volatile("s_wait_asynccnt 0x0" ::: "memory");
}

// ---------------- K0: query f32 -> bf16 (hardware cvt) ----------------
__global__ void cvt_query_kernel(const float* __restrict__ q,
                                 __bf16* __restrict__ qbf) {
    int i = blockIdx.x * 256 + threadIdx.x;
    if (i < BATCH * DIM) qbf[i] = (__bf16)q[i];
}

// ---------------- K1: scores = query @ keysT via bf16 WMMA ----------------
// grid = MEMSZ/128 blocks, 256 threads (8 waves). Block tile: 128 keys x 256 queries.
// Wave w owns key strip [kb + 16w, kb + 16w + 16) x all 16 query strips.
// A (256 bf16 query rows) staged K-half at a time in 132 KB padded LDS via async DMA;
// the inner 8-step K loop runs barrier-free.
__global__ __launch_bounds__(256)
void gemm_scores_kernel(const __bf16* __restrict__ qbf,  // [BATCH][DIM] bf16
                        const float* __restrict__ skey,  // [MEMSZ][DIM] f32
                        float* __restrict__ scores)      // [BATCH][MEMSZ] f32
{
    __shared__ uint4 Asl[BATCH * AROW_U4];  // 256 rows x (256 bf16 + 16B pad) = 132 KB

    const int tid     = threadIdx.x;
    const int lane    = tid & 31;
    const int wave    = tid >> 5;
    const int lane16  = lane & 15;
    const int halfsel = lane >> 4;        // 0: lanes 0-15, 1: lanes 16-31
    const int kb      = blockIdx.x * 128; // key base for this block
    const int col     = kb + wave * 16 + lane16; // this lane's B column (key row)

    const unsigned lds0 = (unsigned)(size_t)(void*)(&Asl[tid * AROW_U4]);     // staging row
    const char*    gq   = (const char*)(qbf + (size_t)tid * DIM);             // global source row
    const uint4*   Ap   = (const uint4*)Asl + lane16 * AROW_U4 + halfsel * 2; // fragment base
    const float*   krow = skey + (size_t)col * DIM + halfsel * 16;            // B source

    f32x8 acc[16];
#pragma unroll
    for (int i = 0; i < 16; ++i) acc[i] = (f32x8)0.0f;

    // prime the B-operand software pipeline (kk = 0)
    f32x16 traw = __builtin_nontemporal_load((const f32x16*)krow);

    for (int h = 0; h < 2; ++h) {
        __syncthreads(); // previous half's LDS reads complete before overwrite
        // stage this K-half of the query matrix: 512 B/thread async DMA burst
#pragma unroll
        for (int off = 0; off < 512; off += 64)
            async_cp64(lds0 + off, gq + (size_t)h * 512 + off);
        async_wait0();
        __syncthreads();

#pragma unroll 1
        for (int kk2 = 0; kk2 < 256; kk2 += 32) {
            // convert current B raw fragment with packed hw cvt, then immediately
            // issue the NT load for the next K-step (latency hidden by 16 WMMAs)
            BF16Vec bf;
            bf.v = __builtin_convertvector(traw, bf16x16);
            const int knext = h * 256 + kk2 + 32;
            if (knext < DIM)
                traw = __builtin_nontemporal_load((const f32x16*)(krow + knext));

            const int k4 = kk2 >> 3; // uint4 index offset within a row
            // 16 WMMAs in groups of 4: 8 ds_load_b128 in flight per wait,
            // then 4 back-to-back v_wmma issues.
#pragma unroll
            for (int g = 0; g < 4; ++g) {
                BF16Vec f0, f1, f2, f3;
                const int i0 = (4 * g + 0) * 16 * AROW_U4 + k4;
                const int i1 = (4 * g + 1) * 16 * AROW_U4 + k4;
                const int i2 = (4 * g + 2) * 16 * AROW_U4 + k4;
                const int i3 = (4 * g + 3) * 16 * AROW_U4 + k4;
                f0.q[0] = Ap[i0]; f0.q[1] = Ap[i0 + 1];
                f1.q[0] = Ap[i1]; f1.q[1] = Ap[i1 + 1];
                f2.q[0] = Ap[i2]; f2.q[1] = Ap[i2 + 1];
                f3.q[0] = Ap[i3]; f3.q[1] = Ap[i3 + 1];
                acc[4 * g + 0] = __builtin_amdgcn_wmma_f32_16x16x32_bf16(
                    false, f0.v, false, bf.v, (short)0, acc[4 * g + 0], false, false);
                acc[4 * g + 1] = __builtin_amdgcn_wmma_f32_16x16x32_bf16(
                    false, f1.v, false, bf.v, (short)0, acc[4 * g + 1], false, false);
                acc[4 * g + 2] = __builtin_amdgcn_wmma_f32_16x16x32_bf16(
                    false, f2.v, false, bf.v, (short)0, acc[4 * g + 2], false, false);
                acc[4 * g + 3] = __builtin_amdgcn_wmma_f32_16x16x32_bf16(
                    false, f3.v, false, bf.v, (short)0, acc[4 * g + 3], false, false);
            }
        }
    }

    // store: C VGPR v -> query row qs*16 + v (+8 for high lanes), column = lane16 within key strip
#pragma unroll
    for (int qs = 0; qs < 16; ++qs) {
        const int qbase = qs * 16 + halfsel * 8;
        const int c     = kb + wave * 16 + lane16;
#pragma unroll
        for (int v = 0; v < 8; ++v) {
            scores[(size_t)(qbase + v) * MEMSZ + c] = acc[qs][v];
        }
    }
}

// ---------------- K2: per-query exact top-256 + KL mask + hinge ----------------
// grid = BATCH blocks, 256 threads.
__global__ __launch_bounds__(256)
void select_kl_kernel(const float* __restrict__ scores,
                      const float* __restrict__ color_feat,   // [BATCH][CDIM]
                      const float* __restrict__ color_value,  // [MEMSZ][CDIM]
                      float* __restrict__ hinge_out)          // [BATCH]
{
    const int q = blockIdx.x, tid = threadIdx.x;
    const int lane = tid & 31, wave = tid >> 5;

    __shared__ int   hist[2048];
    __shared__ float tie_s[CAP_TIE];
    __shared__ int   tie_i[CAP_TIE];
    __shared__ float sel_s[TOPK];
    __shared__ int   sel_i[TOPK];
    __shared__ float kl_s[TOPK];
    __shared__ float cfs[CDIM];
    __shared__ float red_s[256];
    __shared__ int   red_i[256];
    __shared__ int   s_t, s_cnt_top, s_cnt_tie;
    __shared__ float s_pos;

    for (int i = tid; i < 2048; i += 256) hist[i] = 0;
    for (int c = tid; c < CDIM; c += 256) cfs[c] = color_feat[(size_t)q * CDIM + c] + EPSV;
    if (tid == 0) { s_cnt_top = 0; s_cnt_tie = 0; }
    __syncthreads();

    const float* sq = scores + (size_t)q * MEMSZ;

    // pass 1: 11-bit histogram of order-preserving keys
    for (int j = tid; j < MEMSZ; j += 256)
        atomicAdd(&hist[fkey(sq[j]) >> 21], 1);
    __syncthreads();

    // threshold bucket: smallest (from top) where cumulative count reaches TOPK
    if (tid == 0) {
        int cum = 0, b = 2047;
        for (; b >= 0; --b) {
            int c = hist[b];
            if (cum + c >= TOPK) break;
            cum += c;
        }
        s_t = b;
    }
    __syncthreads();
    const int t = s_t;

    // pass 2: collect strictly-above (definite) and threshold-bucket (tie) candidates
    for (int j = tid; j < MEMSZ; j += 256) {
        float s = sq[j];
        int b = (int)(fkey(s) >> 21);
        if (b > t) {
            int p = atomicAdd(&s_cnt_top, 1);
            if (p < TOPK) { sel_s[p] = s; sel_i[p] = j; }
        } else if (b == t) {
            int p = atomicAdd(&s_cnt_tie, 1);
            if (p < CAP_TIE) { tie_s[p] = s; tie_i[p] = j; }
        }
    }
    __syncthreads();

    const int ntop = min(s_cnt_top, TOPK);
    const int need = TOPK - ntop;
    const int nt   = min(s_cnt_tie, CAP_TIE);

    // resolve ties: extract `need` largest from the tie bucket
    for (int r = 0; r < need; ++r) {
        float best = -3.0e38f; int bi = -1;
        for (int j = tid; j < nt; j += 256) {
            float v = tie_s[j];
            if (v > best) { best = v; bi = j; }
        }
        red_s[tid] = best; red_i[tid] = bi;
        __syncthreads();
        for (int st = 128; st > 0; st >>= 1) {
            if (tid < st && red_s[tid + st] > red_s[tid]) {
                red_s[tid] = red_s[tid + st];
                red_i[tid] = red_i[tid + st];
            }
            __syncthreads();
        }
        if (tid == 0) {
            int j = red_i[0];
            if (j >= 0) { sel_s[ntop + r] = red_s[0]; sel_i[ntop + r] = tie_i[j]; tie_s[j] = -3.0e38f; }
            else        { sel_s[ntop + r] = -3.0e38f; sel_i[ntop + r] = 0; }
        }
        __syncthreads();
    }

    // KL per selected row, coalesced: each wave owns 32 rows; for one row all
    // 32 lanes read consecutive elements, then shfl-reduce.
    for (int r = 0; r < 32; ++r) {
        const int slot = wave * 32 + r;
        const int row  = sel_i[slot];
        const float* rv = color_value + (size_t)row * CDIM;
        float part = 0.0f;
        for (int c = lane; c < CDIM; c += 32) {
            float a = rv[c];
            part += a * log10f(a / cfs[c]);
        }
#pragma unroll
        for (int m = 16; m > 0; m >>= 1) part += __shfl_xor(part, m);
        if (lane == 0) kl_s[slot] = part;
    }
    __syncthreads();

    // thread tid owns selected slot tid
    const float s    = sel_s[tid];
    const bool  mask = (kl_s[tid] < THRES);
    float pv = mask ? s : 0.0f;   // == score * mask
    float nv = mask ? 0.0f : s;   // == score * (1 - mask)

    // max-reduce pos, then neg
    red_s[tid] = pv; __syncthreads();
    for (int st = 128; st > 0; st >>= 1) {
        if (tid < st) red_s[tid] = fmaxf(red_s[tid], red_s[tid + st]);
        __syncthreads();
    }
    if (tid == 0) s_pos = red_s[0];
    __syncthreads();

    red_s[tid] = nv; __syncthreads();
    for (int st = 128; st > 0; st >>= 1) {
        if (tid < st) red_s[tid] = fmaxf(red_s[tid], red_s[tid + st]);
        __syncthreads();
    }
    if (tid == 0) hinge_out[q] = fmaxf(red_s[0] - s_pos + ALPHA, 0.0f);
}

// ---------------- K3: mean over BATCH hinges ----------------
__global__ void mean_kernel(const float* __restrict__ h, float* __restrict__ out) {
    __shared__ float r[256];
    int tid = threadIdx.x;
    r[tid] = h[tid];
    __syncthreads();
    for (int st = 128; st > 0; st >>= 1) {
        if (tid < st) r[tid] += r[tid + st];
        __syncthreads();
    }
    if (tid == 0) out[0] = r[0] / (float)BATCH;
}

extern "C" void kernel_launch(void* const* d_in, const int* in_sizes, int n_in,
                              void* d_out, int out_size, void* d_ws, size_t ws_size,
                              hipStream_t stream) {
    const float* query      = (const float*)d_in[0]; // [BATCH][DIM]
    const float* color_feat = (const float*)d_in[1]; // [BATCH][CDIM]
    const float* skey       = (const float*)d_in[2]; // [MEMSZ][DIM]
    const float* cvalue     = (const float*)d_in[3]; // [MEMSZ][CDIM]

    const size_t scores_bytes = (size_t)BATCH * MEMSZ * sizeof(float);      // 256 MB
    const size_t qbf_bytes    = (size_t)BATCH * DIM * sizeof(__bf16);
    const size_t need         = scores_bytes + qbf_bytes + BATCH * sizeof(float);
    if (ws_size < need) return;

    char* ws = (char*)d_ws;
    float*   scores = (float*)ws;
    __bf16*  qbf    = (__bf16*)(ws + scores_bytes);
    float*   hinge  = (float*)(ws + scores_bytes + qbf_bytes);

    cvt_query_kernel<<<(BATCH * DIM + 255) / 256, 256, 0, stream>>>(query, qbf);
    gemm_scores_kernel<<<MEMSZ / 128, 256, 0, stream>>>(qbf, skey, scores);
    select_kl_kernel<<<BATCH, 256, 0, stream>>>(scores, color_feat, cvalue, hinge);
    mean_kernel<<<1, 256, 0, stream>>>(hinge, (float*)d_out);
}
